// Vit2d_13116830122251
// MI455X (gfx1250) — matrix operand order repaired and checked by reading, not compiled, not run on hardware
//
#include <hip/hip_runtime.h>
#include <hip/hip_bf16.h>

// ---------------- problem constants (from reference) ----------------
constexpr int Bc    = 32;
constexpr int Cc    = 3;
constexpr int IMGc  = 256;
constexpr int Pc    = 16;
constexpr int Ec    = 512;
constexpr int HEADSc= 8;
constexpr int Lc    = 6;
constexpr int NYc   = IMGc / Pc;          // 16
constexpr int NXc   = IMGc / Pc;          // 16
constexpr int Nc    = NYc * NXc;          // 256 tokens
constexpr int HCc   = Ec / HEADSc;        // 64
constexpr int TABLEc= (2*NYc-1)*(2*NXc-1);// 961
constexpr int KPEc  = Cc * Pc * Pc;       // 768 (patch-embed K)
constexpr int LSTR  = 20;                 // LDS row stride in dwords (16B aligned, bank-spread)
#define SCALEc 0.125f                      // HC^-0.5

// ---------------- WMMA vector types ----------------
typedef __bf16 v16bf __attribute__((ext_vector_type(16)));
typedef float  v8f   __attribute__((ext_vector_type(8)));

union FragU { v16bf v; uint4 q[2]; unsigned u[8]; };

// ---------------- helpers ----------------
__device__ __forceinline__ unsigned short bf16_rne(float f) {
    union { float f; unsigned u; } v; v.f = f;
    unsigned u = v.u;
    u += 0x7FFFu + ((u >> 16) & 1u);      // round to nearest even
    return (unsigned short)(u >> 16);
}
__device__ __forceinline__ unsigned pack2(unsigned short lo, unsigned short hi) {
    return (unsigned)lo | ((unsigned)hi << 16);
}
__device__ __forceinline__ float gelu_exact(float x) {
    return 0.5f * x * (1.0f + erff(x * 0.70710678118654752f));
}

// A fragment (16x32 bf16). LDS layout ldsA[row][kpair], stride LSTR dwords.
__device__ __forceinline__ v16bf load_fragA(const unsigned* ldsA, int row, int g) {
    FragU f;
    f.q[0] = *(const uint4*)(ldsA + row * LSTR + g * 4);
    f.q[1] = *(const uint4*)(ldsA + row * LSTR + 8 + g * 4);
    return f.v;
}
// B fragment (32x16 bf16). LDS layout ldsB[col][kpair], stride LSTR dwords.
__device__ __forceinline__ v16bf load_fragB(const unsigned* ldsB, int col, int g) {
    FragU f;
    f.q[0] = *(const uint4*)(ldsB + col * LSTR + g * 4);
    f.q[1] = *(const uint4*)(ldsB + col * LSTR + 8 + g * 4);
    return f.v;
}
__device__ __forceinline__ v8f wmma_bf16(v16bf a, v16bf b, v8f c) {
    return __builtin_amdgcn_wmma_f32_16x16x32_bf16(false, a, false, b, (short)0, c, false, false);
}

// ==================================================================
// Generic GEMM: Y[b] = W(OxK, f32) * X[b](KxN, bf16) + bias, WMMA bf16.
// Tile 128x128 per block (4 waves, 2x2 of 64x64 each: 16 WMMA / K-step).
// EPI 0: Ybf = bf16(acc+bias)
// EPI 1: Ybf = bf16(gelu(acc+bias))
// EPI 2: Yf32 += acc+bias (residual), Ybf = bf16(Yf32)
// grid: (N/128, O/128, batch), block: 128
// ==================================================================
template<int EPI>
__global__ __launch_bounds__(128) void gemm_bf16_kernel(
    const float* __restrict__ W, const unsigned short* __restrict__ Xb,
    const float* __restrict__ bias, unsigned short* __restrict__ Ybf,
    float* __restrict__ Yf32, int O, int K, int Nn)
{
    __shared__ __align__(16) unsigned ldsA[128 * LSTR];
    __shared__ __align__(16) unsigned ldsB[128 * LSTR];
    const int tileN = blockIdx.x * 128, tileO = blockIdx.y * 128;
    const size_t bofX = (size_t)blockIdx.z * K * Nn;
    const size_t bofY = (size_t)blockIdx.z * O * Nn;
    const int lane = threadIdx.x & 31, wave = threadIdx.x >> 5;
    const int l16 = lane & 15, g = lane >> 4;
    const int mBase = (wave >> 1) * 64, nBase = (wave & 1) * 64;

    v8f acc[4][4];
#pragma unroll
    for (int i = 0; i < 4; ++i)
#pragma unroll
        for (int j = 0; j < 4; ++j) acc[i][j] = (v8f){0,0,0,0,0,0,0,0};

    for (int kb = 0; kb < K; kb += 32) {
        __syncthreads();
        // stage A (weights f32 -> bf16 pairs): 128 rows x 16 kpairs
        for (int w = threadIdx.x; w < 2048; w += 128) {
            int r = w >> 4, cp = w & 15;
            const float* src = W + (size_t)(tileO + r) * K + kb + 2 * cp;
            ldsA[r * LSTR + cp] = pack2(bf16_rne(src[0]), bf16_rne(src[1]));
        }
        // stage B transposed to [col][kpair]: coalesced over n
        for (int w = threadIdx.x; w < 2048; w += 128) {
            int kp = w >> 7, n = w & 127;
            size_t base = bofX + (size_t)(kb + 2 * kp) * Nn + tileN + n;
            ldsB[n * LSTR + kp] = pack2(Xb[base], Xb[base + Nn]);
        }
        if (kb + 32 < K)
            __builtin_prefetch(&Xb[bofX + (size_t)(kb + 32) * Nn + tileN], 0, 1);
        __syncthreads();

        v16bf fb[4];
#pragma unroll
        for (int j = 0; j < 4; ++j)
            fb[j] = load_fragB(ldsB, nBase + j * 16 + l16, g);
#pragma unroll
        for (int mt = 0; mt < 4; ++mt) {
            v16bf fa = load_fragA(ldsA, mBase + mt * 16 + l16, g);
#pragma unroll
            for (int j = 0; j < 4; ++j)
                acc[mt][j] = wmma_bf16(fa, fb[j], acc[mt][j]);
        }
    }
    // epilogue (C layout: row = g*8 + vgpr, col = lane%16)
#pragma unroll
    for (int mt = 0; mt < 4; ++mt)
#pragma unroll
    for (int j = 0; j < 4; ++j)
#pragma unroll
    for (int r = 0; r < 8; ++r) {
        int m = tileO + mBase + mt * 16 + g * 8 + r;
        int n = tileN + nBase + j * 16 + l16;
        size_t idx = bofY + (size_t)m * Nn + n;
        float v = acc[mt][j][r] + bias[m];
        if (EPI == 1) v = gelu_exact(v);
        if (EPI == 2) {
            float f = Yf32[idx] + v;
            Yf32[idx] = f;
            Ybf[idx] = bf16_rne(f);
        } else {
            Ybf[idx] = bf16_rne(v);
        }
    }
}

// ==================================================================
// Patch embedding as WMMA GEMM with im2col gather on B side.
// grid: (N/128=2, E/128=4, B), block 128
// ==================================================================
__global__ __launch_bounds__(128) void patch_embed_kernel(
    const float* __restrict__ x, const float* __restrict__ ew,
    const float* __restrict__ eb, const float* __restrict__ pos,
    float* __restrict__ hF, unsigned short* __restrict__ hBf)
{
    __shared__ __align__(16) unsigned ldsA[128 * LSTR];
    __shared__ __align__(16) unsigned ldsB[128 * LSTR];
    const int tileN = blockIdx.x * 128, tileO = blockIdx.y * 128;
    const int b = blockIdx.z;
    const int lane = threadIdx.x & 31, wave = threadIdx.x >> 5;
    const int l16 = lane & 15, g = lane >> 4;
    const int mBase = (wave >> 1) * 64, nBase = (wave & 1) * 64;

    v8f acc[4][4];
#pragma unroll
    for (int i = 0; i < 4; ++i)
#pragma unroll
        for (int j = 0; j < 4; ++j) acc[i][j] = (v8f){0,0,0,0,0,0,0,0};

    for (int kb = 0; kb < KPEc; kb += 32) {
        __syncthreads();
        for (int w = threadIdx.x; w < 2048; w += 128) {
            int r = w >> 4, cp = w & 15;
            const float* src = ew + (size_t)(tileO + r) * KPEc + kb + 2 * cp;
            ldsA[r * LSTR + cp] = pack2(bf16_rne(src[0]), bf16_rne(src[1]));
        }
        for (int w = threadIdx.x; w < 2048; w += 128) {
            int kp = w >> 7, n = w & 127;
            int k  = kb + 2 * kp;                 // even; k and k+1 share (c,p) row
            int c  = k >> 8, rem = k & 255;
            int p  = rem >> 4, q = rem & 15;
            int pn = tileN + n;                   // patch index
            int y  = pn >> 4, x0 = pn & 15;
            const float* src = x + (((size_t)b * Cc + c) * IMGc + y * Pc + p) * IMGc
                                 + x0 * Pc + q;
            ldsB[n * LSTR + kp] = pack2(bf16_rne(src[0]), bf16_rne(src[1]));
        }
        __syncthreads();

        v16bf fb[4];
#pragma unroll
        for (int j = 0; j < 4; ++j)
            fb[j] = load_fragB(ldsB, nBase + j * 16 + l16, g);
#pragma unroll
        for (int mt = 0; mt < 4; ++mt) {
            v16bf fa = load_fragA(ldsA, mBase + mt * 16 + l16, g);
#pragma unroll
            for (int j = 0; j < 4; ++j)
                acc[mt][j] = wmma_bf16(fa, fb[j], acc[mt][j]);
        }
    }
#pragma unroll
    for (int mt = 0; mt < 4; ++mt)
#pragma unroll
    for (int j = 0; j < 4; ++j)
#pragma unroll
    for (int r = 0; r < 8; ++r) {
        int m = tileO + mBase + mt * 16 + g * 8 + r;   // e
        int n = tileN + nBase + j * 16 + l16;          // patch
        float v = acc[mt][j][r] + eb[m] + pos[m * Nc + n];
        size_t idx = ((size_t)b * Ec + m) * Nc + n;
        hF[idx]  = v;
        hBf[idx] = bf16_rne(v);
    }
}

// ==================================================================
// Attention scores: att[bh,i,j] = sum_c k[c,i] * q[c,j]   (K^T Q)
// Tile 128x128 (i x j), K = HC = 64. grid: (2, 2, B*HEADS), block 128
// ==================================================================
__global__ __launch_bounds__(128) void attn_scores_kernel(
    const unsigned short* __restrict__ kbuf, const unsigned short* __restrict__ qbuf,
    float* __restrict__ att)
{
    __shared__ __align__(16) unsigned ldsA[128 * LSTR];
    __shared__ __align__(16) unsigned ldsB[128 * LSTR];
    const int bz = blockIdx.z;
    const int b = bz >> 3, h = bz & 7;
    const size_t boff = (size_t)b * Ec * Nc;
    const int hro = h * HCc;
    const int tileJ = blockIdx.x * 128, tileI = blockIdx.y * 128;
    const int lane = threadIdx.x & 31, wave = threadIdx.x >> 5;
    const int l16 = lane & 15, g = lane >> 4;
    const int mBase = (wave >> 1) * 64, nBase = (wave & 1) * 64;

    v8f acc[4][4];
#pragma unroll
    for (int i = 0; i < 4; ++i)
#pragma unroll
        for (int j = 0; j < 4; ++j) acc[i][j] = (v8f){0,0,0,0,0,0,0,0};

    for (int kb = 0; kb < HCc; kb += 32) {
        __syncthreads();
        // A[m=i][k=c] = k[c,i]: coalesced over i
        for (int w = threadIdx.x; w < 2048; w += 128) {
            int cp = w >> 7, r = w & 127, c = kb + 2 * cp;
            unsigned short a0 = kbuf[boff + (size_t)(hro + c) * Nc + tileI + r];
            unsigned short a1 = kbuf[boff + (size_t)(hro + c + 1) * Nc + tileI + r];
            ldsA[r * LSTR + cp] = pack2(a0, a1);
        }
        // B[k=c][n=j] = q[c,j]: coalesced over j
        for (int w = threadIdx.x; w < 2048; w += 128) {
            int kp = w >> 7, n = w & 127, c = kb + 2 * kp;
            unsigned b0 = qbuf[boff + (size_t)(hro + c) * Nc + tileJ + n];
            unsigned b1 = qbuf[boff + (size_t)(hro + c + 1) * Nc + tileJ + n];
            ldsB[n * LSTR + kp] = b0 | (b1 << 16);
        }
        __syncthreads();

        v16bf fb[4];
#pragma unroll
        for (int j = 0; j < 4; ++j)
            fb[j] = load_fragB(ldsB, nBase + j * 16 + l16, g);
#pragma unroll
        for (int mt = 0; mt < 4; ++mt) {
            v16bf fa = load_fragA(ldsA, mBase + mt * 16 + l16, g);
#pragma unroll
            for (int j = 0; j < 4; ++j)
                acc[mt][j] = wmma_bf16(fa, fb[j], acc[mt][j]);
        }
    }
#pragma unroll
    for (int mt = 0; mt < 4; ++mt)
#pragma unroll
    for (int j = 0; j < 4; ++j)
#pragma unroll
    for (int r = 0; r < 8; ++r) {
        int mi = tileI + mBase + mt * 16 + g * 8 + r;
        int nj = tileJ + nBase + j * 16 + l16;
        att[(size_t)bz * Nc * Nc + (size_t)mi * Nc + nj] = acc[mt][j][r];
    }
}

// ==================================================================
// Softmax over keys (axis i) of att*SCALE + pos_enc[h][rel_idx[i,j]]
// grid: B*HEADS blocks, 256 threads (one column j each)
// ==================================================================
__global__ __launch_bounds__(256) void softmax_rel_kernel(
    const float* __restrict__ att, const float* __restrict__ pe_layer,
    const int* __restrict__ rel_idx, unsigned short* __restrict__ attb)
{
    const int bz = blockIdx.x;
    const int h = bz & 7;
    const int j = threadIdx.x;
    const size_t base = (size_t)bz * Nc * Nc;
    const float* peh = pe_layer + (size_t)h * TABLEc;

    float mx = -1e30f;
    for (int i = 0; i < Nc; ++i) {
        float v = att[base + (size_t)i * Nc + j] * SCALEc + peh[rel_idx[i * Nc + j]];
        mx = fmaxf(mx, v);
    }
    float s = 0.f;
    for (int i = 0; i < Nc; ++i) {
        float v = att[base + (size_t)i * Nc + j] * SCALEc + peh[rel_idx[i * Nc + j]];
        s += __expf(v - mx);
    }
    float inv = 1.f / s;
    for (int i = 0; i < Nc; ++i) {
        float v = att[base + (size_t)i * Nc + j] * SCALEc + peh[rel_idx[i * Nc + j]];
        attb[base + (size_t)i * Nc + j] = bf16_rne(__expf(v - mx) * inv);
    }
}

// ==================================================================
// AV: out[c,j] = sum_i v[c,i] * att[i,j];  tile 64 x 128 per block,
// 4 waves along j (each 64x32: 8 WMMA / K-step). grid: (2, 1, B*HEADS)
// A tile is a pure byte copy (v rows are contiguous over i), so it is
// staged with the CDNA5 async-to-LDS engine (ASYNCcnt), overlapping the
// VALU-packed B staging; s_wait_asynccnt 0 before the block barrier.
// ==================================================================
__global__ __launch_bounds__(128) void attn_av_kernel(
    const unsigned short* __restrict__ vbuf, const unsigned short* __restrict__ attb,
    unsigned short* __restrict__ aout)
{
    __shared__ __align__(16) unsigned ldsA[64 * LSTR];
    __shared__ __align__(16) unsigned ldsB[128 * LSTR];
    const int bz = blockIdx.z;
    const int b = bz >> 3, h = bz & 7;
    const size_t boff = (size_t)b * Ec * Nc;
    const size_t abase = (size_t)bz * Nc * Nc;
    const int hro = h * HCc;
    const int tileJ = blockIdx.x * 128;
    const int lane = threadIdx.x & 31, wave = threadIdx.x >> 5;
    const int l16 = lane & 15, g = lane >> 4;
    const int nBase = wave * 32;

    v8f acc[4][2];
#pragma unroll
    for (int i = 0; i < 4; ++i)
#pragma unroll
        for (int j = 0; j < 2; ++j) acc[i][j] = (v8f){0,0,0,0,0,0,0,0};

    for (int kb = 0; kb < Nc; kb += 32) {
        __syncthreads();
        // A[m=c][k=i] = v[c,i]: straight 64B-per-row copy -> async to LDS.
        // 64 rows x 4 x 16B chunks = 256 chunks, 2 per thread.
        for (int w = threadIdx.x; w < 256; w += 128) {
            int r = w >> 2, ch = w & 3;
            const unsigned short* gsrc =
                vbuf + boff + (size_t)(hro + r) * Nc + kb + ch * 8;
            unsigned ldsoff = (unsigned)(size_t)(ldsA + r * LSTR) + ch * 16;
            asm volatile("global_load_async_to_lds_b128 %0, %1, off"
                         :: "v"(ldsoff), "v"(gsrc) : "memory");
        }
        // B[k=i][n=j] = att[i,j] (needs (i,i+1) pair packing -> VALU path)
        for (int w = threadIdx.x; w < 2048; w += 128) {
            int kp = w >> 7, n = w & 127, i = kb + 2 * kp;
            unsigned b0 = attb[abase + (size_t)i * Nc + tileJ + n];
            unsigned b1 = attb[abase + (size_t)(i + 1) * Nc + tileJ + n];
            ldsB[n * LSTR + kp] = b0 | (b1 << 16);
        }
        asm volatile("s_wait_asynccnt 0x0" ::: "memory");
        __syncthreads();

        v16bf fb[2];
#pragma unroll
        for (int j = 0; j < 2; ++j)
            fb[j] = load_fragB(ldsB, nBase + j * 16 + l16, g);
#pragma unroll
        for (int mt = 0; mt < 4; ++mt) {
            v16bf fa = load_fragA(ldsA, mt * 16 + l16, g);
#pragma unroll
            for (int j = 0; j < 2; ++j)
                acc[mt][j] = wmma_bf16(fa, fb[j], acc[mt][j]);
        }
    }
#pragma unroll
    for (int mt = 0; mt < 4; ++mt)
#pragma unroll
    for (int j = 0; j < 2; ++j)
#pragma unroll
    for (int r = 0; r < 8; ++r) {
        int m = mt * 16 + g * 8 + r;                   // c in 0..63
        int n = tileJ + nBase + j * 16 + l16;
        aout[boff + (size_t)(hro + m) * Nc + n] = bf16_rne(acc[mt][j][r]);
    }
}

// ==================================================================
extern "C" void kernel_launch(void* const* d_in, const int* in_sizes, int n_in,
                              void* d_out, int out_size, void* d_ws, size_t ws_size,
                              hipStream_t stream) {
    (void)in_sizes; (void)n_in; (void)out_size; (void)ws_size;
    const float* x      = (const float*)d_in[0];
    const float* emb_w  = (const float*)d_in[1];
    const float* emb_b  = (const float*)d_in[2];
    const float* pos    = (const float*)d_in[3];
    const float* wq     = (const float*)d_in[4];
    const float* bq     = (const float*)d_in[5];
    const float* wk     = (const float*)d_in[6];
    const float* bk     = (const float*)d_in[7];
    const float* wv     = (const float*)d_in[8];
    const float* bv     = (const float*)d_in[9];
    const float* wu     = (const float*)d_in[10];
    const float* bu     = (const float*)d_in[11];
    const float* pe     = (const float*)d_in[12];
    const float* w1     = (const float*)d_in[13];
    const float* b1     = (const float*)d_in[14];
    const float* w2     = (const float*)d_in[15];
    const float* b2     = (const float*)d_in[16];
    const int*   relidx = (const int*)d_in[17];

    float* hF = (float*)d_out;                 // residual stream lives in d_out

    uint8_t* ws = (uint8_t*)d_ws;
    size_t off = 0;
    auto alloc = [&](size_t bytes) -> void* {
        void* p = ws + off;
        off += (bytes + 255) & ~(size_t)255;
        return p;
    };
    const size_t actE  = (size_t)Bc * Ec * Nc;       // elements per (B,E,N)
    unsigned short* hBf  = (unsigned short*)alloc(actE * 2);
    unsigned short* qbuf = (unsigned short*)alloc(actE * 2);
    unsigned short* kbuf = (unsigned short*)alloc(actE * 2);
    unsigned short* vbuf = (unsigned short*)alloc(actE * 2);
    float*          attf = (float*)alloc((size_t)Bc * HEADSc * Nc * Nc * 4);
    unsigned short* attb = (unsigned short*)alloc((size_t)Bc * HEADSc * Nc * Nc * 2);
    unsigned short* aout = (unsigned short*)alloc(actE * 2);
    unsigned short* zbuf = (unsigned short*)alloc((size_t)Bc * 4 * Ec * Nc * 2);

    // 1) patch embedding (WMMA GEMM, K=768) -> hF (f32) + hBf (bf16)
    patch_embed_kernel<<<dim3(Nc / 128, Ec / 128, Bc), 128, 0, stream>>>(
        x, emb_w, emb_b, pos, hF, hBf);

    const dim3 gProj(Nc / 128, Ec / 128, Bc);          // (2, 4, 32)
    const dim3 gMlp1(Nc / 128, (4 * Ec) / 128, Bc);    // (2, 16, 32)
    const dim3 gAtt(Nc / 128, Nc / 128, Bc * HEADSc);  // (2, 2, 256)
    const dim3 gAV(Nc / 128, 1, Bc * HEADSc);          // (2, 1, 256)

    for (int i = 0; i < Lc; ++i) {
        const float* wq_i = wq + (size_t)i * Ec * Ec;
        const float* wk_i = wk + (size_t)i * Ec * Ec;
        const float* wv_i = wv + (size_t)i * Ec * Ec;
        const float* wu_i = wu + (size_t)i * Ec * Ec;
        const float* w1_i = w1 + (size_t)i * 4 * Ec * Ec;
        const float* w2_i = w2 + (size_t)i * 4 * Ec * Ec;
        const float* bq_i = bq + (size_t)i * Ec;
        const float* bk_i = bk + (size_t)i * Ec;
        const float* bv_i = bv + (size_t)i * Ec;
        const float* bu_i = bu + (size_t)i * Ec;
        const float* b1_i = b1 + (size_t)i * 4 * Ec;
        const float* b2_i = b2 + (size_t)i * Ec;
        const float* pe_i = pe + (size_t)i * HEADSc * TABLEc;

        // q, k, v projections (bf16 outputs)
        gemm_bf16_kernel<0><<<gProj, 128, 0, stream>>>(wq_i, hBf, bq_i, qbuf, (float*)nullptr, Ec, Ec, Nc);
        gemm_bf16_kernel<0><<<gProj, 128, 0, stream>>>(wk_i, hBf, bk_i, kbuf, (float*)nullptr, Ec, Ec, Nc);
        gemm_bf16_kernel<0><<<gProj, 128, 0, stream>>>(wv_i, hBf, bv_i, vbuf, (float*)nullptr, Ec, Ec, Nc);

        // attention
        attn_scores_kernel<<<gAtt, 128, 0, stream>>>(kbuf, qbuf, attf);
        softmax_rel_kernel<<<Bc * HEADSc, 256, 0, stream>>>(attf, pe_i, relidx, attb);
        attn_av_kernel<<<gAV, 128, 0, stream>>>(vbuf, attb, aout);

        // output projection + residual into hF / hBf
        gemm_bf16_kernel<2><<<gProj, 128, 0, stream>>>(wu_i, aout, bu_i, hBf, hF, Ec, Ec, Nc);

        // MLP: gelu(w1 h + b1) -> z ; h += w2 z + b2
        gemm_bf16_kernel<1><<<gMlp1, 128, 0, stream>>>(w1_i, hBf, b1_i, zbuf, (float*)nullptr, 4 * Ec, Ec, Nc);
        gemm_bf16_kernel<2><<<gProj, 128, 0, stream>>>(w2_i, zbuf, b2_i, hBf, hF, Ec, 4 * Ec, Nc);
    }
}
